// Transformer_7387343749821
// MI455X (gfx1250) — matrix operand order repaired and checked by reading, not compiled, not run on hardware
//
#include <hip/hip_runtime.h>
#include <cmath>

#define PSZ    16
#define IMG    800
#define GRIDW  50
#define NP     2500
#define DMODEL 768
#define NH     8
#define HDIM   96
#define FFDIM  2048
#define NL     2
#define BATCH  2
#define NV     1250

typedef __bf16 bf16;
typedef __attribute__((ext_vector_type(16))) __bf16 v16bf;
typedef __attribute__((ext_vector_type(8)))  float  v8f;

__device__ inline v8f wmma_bf16(v16bf a, v16bf b, v8f c) {
    // D = A(16x32 bf16) * B(32x16 bf16) + C(16x16 f32)
    return __builtin_amdgcn_wmma_f32_16x16x32_bf16(
        /*neg_a=*/false, a, /*neg_b=*/false, b,
        /*c_mod=*/(short)0, c, /*reuse_a=*/false, /*reuse_b=*/false);
}

// ---------------------------------------------------------------------------
// fp32 -> bf16 conversion (weights)
// ---------------------------------------------------------------------------
__global__ __launch_bounds__(256) void f2bf(const float* __restrict__ s,
                                            bf16* __restrict__ d, int n) {
    int i = blockIdx.x * 256 + threadIdx.x;
    if (i < n) d[i] = (bf16)s[i];
}

// ---------------------------------------------------------------------------
// patchify + gather visible + add enc_pos ; out [B*NV, D]
// ---------------------------------------------------------------------------
__global__ __launch_bounds__(256) void patchify_gather(
    const float* __restrict__ x, const int* __restrict__ vis_idx,
    const float* __restrict__ enc_pos, float* __restrict__ out) {
    int row = blockIdx.x;              // b*NV + j
    int b = row / NV, j = row % NV;
    int p = vis_idx[b * NV + j];
    int gy = p / GRIDW, gx = p % GRIDW;
    for (int d = threadIdx.x; d < DMODEL; d += 256) {
        int c = d >> 8, py = (d >> 4) & 15, px = d & 15;
        float v = x[((size_t)(b * 3 + c) * IMG + gy * PSZ + py) * IMG + gx * PSZ + px];
        out[(size_t)row * DMODEL + d] = v + enc_pos[(size_t)j * DMODEL + d];
    }
}

// ---------------------------------------------------------------------------
// GEMM: C[M,N] = act(A[M,K]_f32 * W[K,N]_bf16 + bias[N])
// block tile 128x128, 8 waves (4x2), wave tile 32x64 -> 8 WMMA / K-step.
// Double-buffered LDS; B staged in fragment-ready swizzle [khalf][t][lane][8].
// ---------------------------------------------------------------------------
template <int ACT>
__global__ __launch_bounds__(256) void gemm_bias(
    const float* __restrict__ A, const bf16* __restrict__ W,
    const float* __restrict__ bias, float* __restrict__ C,
    int M, int K, int N) {
    __shared__ bf16 As[2][128 * 40];     // 128x32, padded row stride 40
    __shared__ bf16 Bs[2][4096];         // [khalf(2)][ntile(8)][lane(32)][8]

    const int tid  = threadIdx.x;
    const int lane = tid & 31;
    const int w    = tid >> 5;      // 0..7
    const int wm   = w >> 1;        // 0..3  (M strip of 32)
    const int wn   = w & 1;         // 0..1  (N strip of 64)
    const int mBase = blockIdx.y * 128;
    const int nBase = blockIdx.x * 128;

    v8f acc[2][4];
#pragma unroll
    for (int i = 0; i < 2; ++i)
#pragma unroll
        for (int j = 0; j < 4; ++j)
#pragma unroll
            for (int e = 0; e < 8; ++e) acc[i][j][e] = 0.f;

    auto stageA = [&](int kt, int buf) {
#pragma unroll
        for (int i = 0; i < 16; ++i) {
            int idx = i * 256 + tid;
            int r = idx >> 5, c = idx & 31;
            int gm = mBase + r;
            float v = (gm < M) ? A[(size_t)gm * K + kt + c] : 0.f;
            As[buf][r * 40 + c] = (bf16)v;
        }
    };
    auto stageB = [&](int kt, int buf) {
#pragma unroll
        for (int i = 0; i < 16; ++i) {
            int idx = i * 256 + tid;
            int k = idx >> 7, c = idx & 127;
            bf16 v = W[(size_t)(kt + k) * N + nBase + c];
            int t = c >> 4;
            int ln = (c & 15) | ((k & 16) ? 16 : 0);
            int e = k & 15;
            Bs[buf][((e >> 3) * 8 + t) * 256 + ln * 8 + (e & 7)] = v;
        }
    };

    const int nsteps = K >> 5;
    stageA(0, 0);
    stageB(0, 0);
    __syncthreads();

    for (int s = 0; s < nsteps; ++s) {
        int cur = s & 1;
        if (s + 1 < nsteps) {
            stageA((s + 1) * 32, cur ^ 1);
            stageB((s + 1) * 32, cur ^ 1);
            __builtin_prefetch(&W[(size_t)((s + 1) * 32) * N + nBase + (tid & 127)], 0, 0);
        }

        v16bf af[2], bfr[4];
#pragma unroll
        for (int mi = 0; mi < 2; ++mi) {
            const bf16* ap = &As[cur][(wm * 32 + mi * 16 + (lane & 15)) * 40];
            int kb = (lane & 16) ? 16 : 0;
#pragma unroll
            for (int e = 0; e < 8; ++e) {
                af[mi][e]     = ap[kb + e];
                af[mi][8 + e] = ap[kb + 8 + e];
            }
        }
#pragma unroll
        for (int nt = 0; nt < 4; ++nt) {
            int t = wn * 4 + nt;
            const bf16* p0 = &Bs[cur][t * 256 + lane * 8];
            const bf16* p1 = &Bs[cur][(8 + t) * 256 + lane * 8];
#pragma unroll
            for (int e = 0; e < 8; ++e) {
                bfr[nt][e]     = p0[e];
                bfr[nt][8 + e] = p1[e];
            }
        }
#pragma unroll
        for (int mi = 0; mi < 2; ++mi)
#pragma unroll
            for (int nt = 0; nt < 4; ++nt)
                acc[mi][nt] = wmma_bf16(af[mi], bfr[nt], acc[mi][nt]);
        __syncthreads();
    }

#pragma unroll
    for (int mi = 0; mi < 2; ++mi)
#pragma unroll
        for (int nt = 0; nt < 4; ++nt) {
            int ncol = nBase + wn * 64 + nt * 16 + (lane & 15);
            float bv = bias[ncol];
            int mrow0 = mBase + wm * 32 + mi * 16 + ((lane & 16) ? 8 : 0);
#pragma unroll
            for (int r = 0; r < 8; ++r) {
                int gm = mrow0 + r;
                if (gm < M) {
                    float v = acc[mi][nt][r] + bv;
                    if (ACT) v = fmaxf(v, 0.f);
                    C[(size_t)gm * N + ncol] = v;
                }
            }
        }
}

// ---------------------------------------------------------------------------
// split qkv [M,3D] fp32 -> head-major bf16 q/k/v [B,H,N,HDIM]
// ---------------------------------------------------------------------------
__global__ __launch_bounds__(256) void split_qkv(
    const float* __restrict__ qkv, bf16* __restrict__ q,
    bf16* __restrict__ k, bf16* __restrict__ v, int N) {
    int row = blockIdx.x;            // b*N + n
    int b = row / N, n = row % N;
    const float* src = qkv + (size_t)row * 3 * DMODEL;
    for (int d = threadIdx.x; d < DMODEL; d += 256) {
        int h = d / HDIM, hd = d % HDIM;
        size_t dst = ((size_t)(b * NH + h) * N + n) * HDIM + hd;
        q[dst] = (bf16)src[d];
        k[dst] = (bf16)src[DMODEL + d];
        v[dst] = (bf16)src[2 * DMODEL + d];
    }
}

// ---------------------------------------------------------------------------
// attention: one workgroup = (b, h, 16-query tile). Full score row in LDS.
// NMT is always even (Npad multiple of 32).
// ---------------------------------------------------------------------------
__global__ __launch_bounds__(256) void attention_kernel(
    const bf16* __restrict__ qm, const bf16* __restrict__ km,
    const bf16* __restrict__ vm, float* __restrict__ ao,
    int N, int NMT, float scale) {
    extern __shared__ char smem[];
    const int Npad = NMT * 16;
    float* Ss = (float*)smem;                                   // 16 x Npad
    bf16* Qs  = (bf16*)(smem + (size_t)16 * Npad * 4);          // 16 x HDIM
    bf16* Vs  = Qs + 16 * HDIM;                                 // 2 x [2][6][32][8]

    const int tid  = threadIdx.x;
    const int lane = tid & 31;
    const int w    = tid >> 5;
    const int q0   = blockIdx.x * 16;
    const int bh   = blockIdx.y;                // b*NH + h
    const int b    = bh / NH, h = bh % NH;
    const size_t base = (size_t)bh * N * HDIM;

    // stage Q tile
    for (int i = tid; i < 16 * HDIM; i += 256) {
        int r = i / HDIM, d = i % HDIM;
        int qr = q0 + r;
        Qs[i] = (qr < N) ? qm[base + (size_t)qr * HDIM + d] : (bf16)0.f;
    }
    __syncthreads();

    // ---- scores: S[16, Npad] = Q * K^T * scale  (8 waves stride m-tiles)
    for (int mt = w; mt < NMT; mt += 8) {
        int m0 = mt * 16;
        int cm = m0 + (lane & 15);
        v8f acc;
#pragma unroll
        for (int e = 0; e < 8; ++e) acc[e] = 0.f;
#pragma unroll
        for (int dk = 0; dk < HDIM; dk += 32) {
            v16bf a, bb;
            int kb = (lane & 16) ? 16 : 0;
            const bf16* ap = &Qs[(lane & 15) * HDIM + dk];
#pragma unroll
            for (int e = 0; e < 8; ++e) {
                a[e]     = ap[kb + e];
                a[8 + e] = ap[kb + 8 + e];
            }
            int kb2 = (lane & 16) ? 16 : 0;
            if (cm < N) {
                bb = *(const v16bf*)(km + base + (size_t)cm * HDIM + dk + kb2);
            } else {
#pragma unroll
                for (int e = 0; e < 16; ++e) bb[e] = (bf16)0.f;
            }
            acc = wmma_bf16(a, bb, acc);
        }
#pragma unroll
        for (int r = 0; r < 8; ++r) {
            int rr = r + ((lane & 16) ? 8 : 0);
            Ss[(size_t)rr * Npad + cm] = (cm < N) ? acc[r] * scale : -1e30f;
        }
    }
    __syncthreads();

    // ---- softmax over m (2 rows per wave, wave32 shuffle reductions)
    for (int r = w * 2; r < w * 2 + 2; ++r) {
        float* row = Ss + (size_t)r * Npad;
        float mx = -3.4e38f;
        for (int m = lane; m < Npad; m += 32) mx = fmaxf(mx, row[m]);
#pragma unroll
        for (int off = 16; off; off >>= 1) mx = fmaxf(mx, __shfl_xor(mx, off, 32));
        float sum = 0.f;
        for (int m = lane; m < Npad; m += 32) {
            float e = __expf(row[m] - mx);
            row[m] = e;
            sum += e;
        }
#pragma unroll
        for (int off = 16; off; off >>= 1) sum += __shfl_xor(sum, off, 32);
        float inv = 1.f / sum;
        for (int m = lane; m < Npad; m += 32) row[m] *= inv;
    }
    __syncthreads();

    // ---- AV: out[16, HDIM] = P * V   (waves 0..5 own 16-wide d tiles)
    // V staged double-buffered in fragment-ready swizzle [khalf][t][lane][8]
    auto stageV = [&](int mb, int buf) {
        bf16* dst = Vs + buf * (2 * 6 * 256);
        for (int i = tid; i < 32 * HDIM; i += 256) {
            int m = i / HDIM, d = i % HDIM;
            int gm = mb + m;
            bf16 val = (gm < N) ? vm[base + (size_t)gm * HDIM + d] : (bf16)0.f;
            int t = d >> 4;
            int ln = (d & 15) | ((m & 16) ? 16 : 0);
            int e = m & 15;
            dst[((e >> 3) * 6 + t) * 256 + ln * 8 + (e & 7)] = val;
        }
    };

    v8f oacc;
#pragma unroll
    for (int e = 0; e < 8; ++e) oacc[e] = 0.f;

    const int msteps = Npad >> 5;
    stageV(0, 0);
    __syncthreads();
    for (int s = 0; s < msteps; ++s) {
        int cur = s & 1;
        if (s + 1 < msteps) stageV((s + 1) * 32, cur ^ 1);
        if (w < 6) {
            int mb = s * 32;
            v16bf a, bb;
            int kb = (lane & 16) ? 16 : 0;
            const float* srow = Ss + (size_t)(lane & 15) * Npad + mb;
#pragma unroll
            for (int e = 0; e < 8; ++e) {
                a[e]     = (bf16)srow[kb + e];
                a[8 + e] = (bf16)srow[kb + 8 + e];
            }
            const bf16* vb = Vs + cur * (2 * 6 * 256);
            const bf16* p0 = &vb[w * 256 + lane * 8];
            const bf16* p1 = &vb[(6 + w) * 256 + lane * 8];
#pragma unroll
            for (int e = 0; e < 8; ++e) {
                bb[e]     = p0[e];
                bb[8 + e] = p1[e];
            }
            oacc = wmma_bf16(a, bb, oacc);
        }
        __syncthreads();
    }
    if (w < 6) {
        int d = h * HDIM + w * 16 + (lane & 15);
        int r0 = (lane & 16) ? 8 : 0;
#pragma unroll
        for (int r = 0; r < 8; ++r) {
            int qr = q0 + r0 + r;
            if (qr < N) ao[((size_t)b * N + qr) * DMODEL + d] = oacc[r];
        }
    }
}

// ---------------------------------------------------------------------------
// x = LN(x + res) * g + be     (one 256-thread block per row, D=768=3*256)
// ---------------------------------------------------------------------------
__global__ __launch_bounds__(256) void ln_residual(
    float* __restrict__ x, const float* __restrict__ res,
    const float* __restrict__ g, const float* __restrict__ be) {
    __shared__ float red1[8];
    __shared__ float red2[8];
    const int row = blockIdx.x;
    const int tid = threadIdx.x;
    const int lane = tid & 31, w = tid >> 5;
    float v[3];
    float s = 0.f;
#pragma unroll
    for (int i = 0; i < 3; ++i) {
        int d = tid + i * 256;
        v[i] = x[(size_t)row * DMODEL + d] + res[(size_t)row * DMODEL + d];
        s += v[i];
    }
#pragma unroll
    for (int off = 16; off; off >>= 1) s += __shfl_xor(s, off, 32);
    if (lane == 0) red1[w] = s;
    __syncthreads();
    float tot = 0.f;
#pragma unroll
    for (int i = 0; i < 8; ++i) tot += red1[i];
    float mean = tot * (1.f / DMODEL);
    float s2 = 0.f;
#pragma unroll
    for (int i = 0; i < 3; ++i) {
        float d = v[i] - mean;
        s2 += d * d;
    }
#pragma unroll
    for (int off = 16; off; off >>= 1) s2 += __shfl_xor(s2, off, 32);
    if (lane == 0) red2[w] = s2;
    __syncthreads();
    float tot2 = 0.f;
#pragma unroll
    for (int i = 0; i < 8; ++i) tot2 += red2[i];
    float rstd = rsqrtf(tot2 * (1.f / DMODEL) + 1e-5f);
#pragma unroll
    for (int i = 0; i < 3; ++i) {
        int d = tid + i * 256;
        x[(size_t)row * DMODEL + d] = (v[i] - mean) * rstd * g[d] + be[d];
    }
}

// ---------------------------------------------------------------------------
// rank of each visible index (== its position after sort)
// ---------------------------------------------------------------------------
__global__ __launch_bounds__(256) void compute_ranks(
    const int* __restrict__ idx, int* __restrict__ ranks) {
    int t = blockIdx.x * 256 + threadIdx.x;
    if (t >= BATCH * NV) return;
    int b = t / NV, j = t % NV;
    const int* p = idx + b * NV;
    int vv = p[j], r = 0;
    for (int k = 0; k < NV; ++k) r += (p[k] < vv) ? 1 : 0;
    ranks[t] = r;
}

// dec_in[b,p,:] = mask_token + dec_pos[p]
__global__ __launch_bounds__(256) void dec_init(
    const float* __restrict__ mask_tok, const float* __restrict__ dec_pos,
    float* __restrict__ out) {
    int row = blockIdx.x;            // b*NP + p
    int p = row % NP;
    for (int d = threadIdx.x; d < DMODEL; d += 256)
        out[(size_t)row * DMODEL + d] = mask_tok[d] + dec_pos[(size_t)p * DMODEL + d];
}

// dec_in[b, idx[j], :] = henc[b, rank[j], :] + dec_pos[idx[j]]
__global__ __launch_bounds__(256) void dec_scatter(
    const float* __restrict__ henc, const int* __restrict__ idx,
    const int* __restrict__ ranks, const float* __restrict__ dec_pos,
    float* __restrict__ out) {
    int t = blockIdx.x;              // b*NV + j
    int b = t / NV;
    int p = idx[t];
    int r = ranks[t];
    for (int d = threadIdx.x; d < DMODEL; d += 256)
        out[((size_t)b * NP + p) * DMODEL + d] =
            henc[((size_t)b * NV + r) * DMODEL + d] + dec_pos[(size_t)p * DMODEL + d];
}

// ---------------------------------------------------------------------------
static inline int ceil_div(int a, int b) { return (a + b - 1) / b; }

extern "C" void kernel_launch(void* const* d_in, const int* in_sizes, int n_in,
                              void* d_out, int out_size, void* d_ws, size_t ws_size,
                              hipStream_t stream) {
    const float* x        = (const float*)d_in[0];
    const int*   vis_idx  = (const int*)d_in[1];
    const float* e_qkv_w  = (const float*)d_in[3];
    const float* e_qkv_b  = (const float*)d_in[4];
    const float* e_out_w  = (const float*)d_in[5];
    const float* e_out_b  = (const float*)d_in[6];
    const float* e_w1     = (const float*)d_in[7];
    const float* e_b1     = (const float*)d_in[8];
    const float* e_w2     = (const float*)d_in[9];
    const float* e_b2     = (const float*)d_in[10];
    const float* e_g1     = (const float*)d_in[11];
    const float* e_be1    = (const float*)d_in[12];
    const float* e_g2     = (const float*)d_in[13];
    const float* e_be2    = (const float*)d_in[14];
    const float* d_qkv_w  = (const float*)d_in[15];
    const float* d_qkv_b  = (const float*)d_in[16];
    const float* d_out_w  = (const float*)d_in[17];
    const float* d_out_b  = (const float*)d_in[18];
    const float* d_w1     = (const float*)d_in[19];
    const float* d_b1     = (const float*)d_in[20];
    const float* d_w2     = (const float*)d_in[21];
    const float* d_b2     = (const float*)d_in[22];
    const float* d_g1     = (const float*)d_in[23];
    const float* d_be1    = (const float*)d_in[24];
    const float* d_g2     = (const float*)d_in[25];
    const float* d_be2    = (const float*)d_in[26];
    const float* out_w    = (const float*)d_in[27];
    const float* out_b    = (const float*)d_in[28];
    const float* mask_tok = (const float*)d_in[29];
    const float* enc_pos  = (const float*)d_in[30];
    const float* dec_pos  = (const float*)d_in[31];

    char* ws = (char*)d_ws;
    size_t off = 0;
    auto carve = [&](size_t bytes) -> char* {
        char* p = ws + off;
        off = (off + bytes + 255) & ~(size_t)255;
        return p;
    };

    bf16* wEncQkv = (bf16*)carve((size_t)NL * DMODEL * 3 * DMODEL * 2);
    bf16* wEncOut = (bf16*)carve((size_t)NL * DMODEL * DMODEL * 2);
    bf16* wEncW1  = (bf16*)carve((size_t)NL * DMODEL * FFDIM * 2);
    bf16* wEncW2  = (bf16*)carve((size_t)NL * FFDIM * DMODEL * 2);
    bf16* wDecQkv = (bf16*)carve((size_t)NL * DMODEL * 3 * DMODEL * 2);
    bf16* wDecOut = (bf16*)carve((size_t)NL * DMODEL * DMODEL * 2);
    bf16* wDecW1  = (bf16*)carve((size_t)NL * DMODEL * FFDIM * 2);
    bf16* wDecW2  = (bf16*)carve((size_t)NL * FFDIM * DMODEL * 2);
    bf16* wFin    = (bf16*)carve((size_t)DMODEL * DMODEL * 2);
    float* hA   = (float*)carve((size_t)BATCH * NP * DMODEL * 4);
    float* hB   = (float*)carve((size_t)BATCH * NP * DMODEL * 4);
    float* tmp1 = (float*)carve((size_t)BATCH * NP * 3 * DMODEL * 4);
    bf16* qB    = (bf16*)carve((size_t)BATCH * NP * DMODEL * 2);
    bf16* kB    = (bf16*)carve((size_t)BATCH * NP * DMODEL * 2);
    bf16* vB    = (bf16*)carve((size_t)BATCH * NP * DMODEL * 2);
    float* aoB  = (float*)carve((size_t)BATCH * NP * DMODEL * 4);
    int* ranks  = (int*)carve((size_t)BATCH * NV * 4);

    auto cvt = [&](const float* src, bf16* dst, size_t n) {
        f2bf<<<dim3((unsigned)((n + 255) / 256)), 256, 0, stream>>>(src, dst, (int)n);
    };
    cvt(e_qkv_w, wEncQkv, (size_t)NL * DMODEL * 3 * DMODEL);
    cvt(e_out_w, wEncOut, (size_t)NL * DMODEL * DMODEL);
    cvt(e_w1,    wEncW1,  (size_t)NL * DMODEL * FFDIM);
    cvt(e_w2,    wEncW2,  (size_t)NL * FFDIM * DMODEL);
    cvt(d_qkv_w, wDecQkv, (size_t)NL * DMODEL * 3 * DMODEL);
    cvt(d_out_w, wDecOut, (size_t)NL * DMODEL * DMODEL);
    cvt(d_w1,    wDecW1,  (size_t)NL * DMODEL * FFDIM);
    cvt(d_w2,    wDecW2,  (size_t)NL * FFDIM * DMODEL);
    cvt(out_w,   wFin,    (size_t)DMODEL * DMODEL);

    const float scale = 1.0f / sqrtf((float)HDIM);

    auto run_block = [&](float* h, int N,
                         const bf16* wq, const float* bq,
                         const bf16* wo, const float* bo,
                         const bf16* w1, const float* b1,
                         const bf16* w2, const float* b2,
                         const float* g1, const float* be1,
                         const float* g2, const float* be2) {
        int M = BATCH * N;
        gemm_bias<0><<<dim3(3 * DMODEL / 128, ceil_div(M, 128)), 256, 0, stream>>>(
            h, wq, bq, tmp1, M, DMODEL, 3 * DMODEL);
        split_qkv<<<dim3(M), 256, 0, stream>>>(tmp1, qB, kB, vB, N);
        int NMT = 2 * ceil_div(N, 32);   // Npad multiple of 32
        size_t smem = (size_t)16 * NMT * 16 * 4    // scores
                    + (size_t)16 * HDIM * 2        // Q tile
                    + (size_t)2 * 2 * 6 * 256 * 2; // V double buffer
        attention_kernel<<<dim3(NMT, BATCH * NH), 256, smem, stream>>>(
            qB, kB, vB, aoB, N, NMT, scale);
        gemm_bias<0><<<dim3(DMODEL / 128, ceil_div(M, 128)), 256, 0, stream>>>(
            aoB, wo, bo, tmp1, M, DMODEL, DMODEL);
        ln_residual<<<dim3(M), 256, 0, stream>>>(h, tmp1, g1, be1);
        gemm_bias<1><<<dim3(FFDIM / 128, ceil_div(M, 128)), 256, 0, stream>>>(
            h, w1, b1, tmp1, M, DMODEL, FFDIM);
        gemm_bias<0><<<dim3(DMODEL / 128, ceil_div(M, 128)), 256, 0, stream>>>(
            tmp1, w2, b2, aoB, M, FFDIM, DMODEL);
        ln_residual<<<dim3(M), 256, 0, stream>>>(h, aoB, g2, be2);
    };

    // ---- encoder on visible patches
    patchify_gather<<<dim3(BATCH * NV), 256, 0, stream>>>(x, vis_idx, enc_pos, hA);
    for (int i = 0; i < NL; ++i) {
        run_block(hA, NV,
                  wEncQkv + (size_t)i * DMODEL * 3 * DMODEL, e_qkv_b + (size_t)i * 3 * DMODEL,
                  wEncOut + (size_t)i * DMODEL * DMODEL,     e_out_b + (size_t)i * DMODEL,
                  wEncW1  + (size_t)i * DMODEL * FFDIM,      e_b1 + (size_t)i * FFDIM,
                  wEncW2  + (size_t)i * FFDIM * DMODEL,      e_b2 + (size_t)i * DMODEL,
                  e_g1 + (size_t)i * DMODEL, e_be1 + (size_t)i * DMODEL,
                  e_g2 + (size_t)i * DMODEL, e_be2 + (size_t)i * DMODEL);
    }

    // ---- scatter into full token set with mask tokens + dec_pos
    compute_ranks<<<dim3(ceil_div(BATCH * NV, 256)), 256, 0, stream>>>(vis_idx, ranks);
    dec_init<<<dim3(BATCH * NP), 256, 0, stream>>>(mask_tok, dec_pos, hB);
    dec_scatter<<<dim3(BATCH * NV), 256, 0, stream>>>(hA, vis_idx, ranks, dec_pos, hB);

    // ---- decoder on all tokens
    for (int i = 0; i < NL; ++i) {
        run_block(hB, NP,
                  wDecQkv + (size_t)i * DMODEL * 3 * DMODEL, d_qkv_b + (size_t)i * 3 * DMODEL,
                  wDecOut + (size_t)i * DMODEL * DMODEL,     d_out_b + (size_t)i * DMODEL,
                  wDecW1  + (size_t)i * DMODEL * FFDIM,      d_b1 + (size_t)i * FFDIM,
                  wDecW2  + (size_t)i * FFDIM * DMODEL,      d_b2 + (size_t)i * DMODEL,
                  d_g1 + (size_t)i * DMODEL, d_be1 + (size_t)i * DMODEL,
                  d_g2 + (size_t)i * DMODEL, d_be2 + (size_t)i * DMODEL);
    }

    // ---- final projection
    gemm_bias<0><<<dim3(DMODEL / 128, ceil_div(BATCH * NP, 128)), 256, 0, stream>>>(
        hB, wFin, out_b, (float*)d_out, BATCH * NP, DMODEL, DMODEL);
}
